// SparseLinear_76656576299096
// MI455X (gfx1250) — matrix operand order repaired and checked by reading, not compiled, not run on hardware
//
#include <hip/hip_runtime.h>

// CDNA5 / gfx1250, wave32. bf16x3-split block-sparse SDD GEMM, double-buffered LDS.

typedef __attribute__((ext_vector_type(16))) __bf16  v16bf;
typedef __attribute__((ext_vector_type(8)))  float   v8f;

#define BLKSZ  128          // output block edge
#define HID    1024         // K dimension
#define BK     32           // K tile = one bf16 WMMA depth
#define BKP    40           // padded LDS row stride (halfs): 80B rows -> 16B aligned
#define NSTAGE (HID / BK)   // 32
#define LDSZ   (BLKSZ * BKP)

union Frag32 { uint4  u[2]; v16bf v; };   // 32-byte WMMA operand from two b128 LDS loads
union Pack4  { __bf16 b[4]; uint2 u; };   // 4 halfs -> one 8-byte LDS store

// hi/lo bf16 split of 4+4 staged floats, stored to one LDS buffer
__device__ __forceinline__ void split_store(const float4* sa, const float4* sw,
                                            __bf16* ahi, __bf16* alo,
                                            __bf16* whi, __bf16* wlo,
                                            const int* loff)
{
#pragma unroll
    for (int i = 0; i < 4; ++i) {
        const float av[4] = {sa[i].x, sa[i].y, sa[i].z, sa[i].w};
        const float wv[4] = {sw[i].x, sw[i].y, sw[i].z, sw[i].w};
        Pack4 ah, al, wh, wl;
#pragma unroll
        for (int e = 0; e < 4; ++e) {
            const __bf16 h = (__bf16)av[e];
            ah.b[e] = h;
            al.b[e] = (__bf16)(av[e] - (float)h);
            const __bf16 hw = (__bf16)wv[e];
            wh.b[e] = hw;
            wl.b[e] = (__bf16)(wv[e] - (float)hw);
        }
        *(uint2*)(&ahi[loff[i]]) = ah.u;
        *(uint2*)(&alo[loff[i]]) = al.u;
        *(uint2*)(&whi[loff[i]]) = wh.u;
        *(uint2*)(&wlo[loff[i]]) = wl.u;
    }
}

__global__ __launch_bounds__(256)
void sdd_bf16x3_kernel(const float* __restrict__ x,
                       const float* __restrict__ w1,
                       const int*   __restrict__ rowi,
                       const int*   __restrict__ coli,
                       float*       __restrict__ out)
{
    // double-buffered: 8 * 10240B = 80KB (<320KB WGP LDS; 4 WGs/WGP possible)
    __shared__ __bf16 Ahi[2][LDSZ];
    __shared__ __bf16 Alo[2][LDSZ];
    __shared__ __bf16 Whi[2][LDSZ];
    __shared__ __bf16 Wlo[2][LDSZ];

    const int nz   = blockIdx.x;
    const int tid  = threadIdx.x;
    const int lane = tid & 31;
    const int wave = tid >> 5;

    const float* Ag = x  + (size_t)rowi[nz] * (BLKSZ * HID);
    const float* Wg = w1 + (size_t)coli[nz] * (BLKSZ * HID);

    // wave -> 32x64 patch of the 128x128 block (2 x 4 tiles of 16x16)
    const int wm = (wave >> 1) * 32;
    const int wn = (wave & 1) * 64;

    // loop-invariant per-thread offsets: global (elements) and LDS (halfs)
    int goff[4], loff[4];
#pragma unroll
    for (int i = 0; i < 4; ++i) {
        const int g   = tid + i * 256;        // 1024 groups of 4 floats
        const int row = g >> 3;
        const int c4  = (g & 7) << 2;
        goff[i] = row * HID + c4;
        loff[i] = row * BKP + c4;
    }

    // loop-invariant fragment LDS offsets (halfs)
    int arow[2], brow[4];
#pragma unroll
    for (int mt = 0; mt < 2; ++mt)
        arow[mt] = (wm + mt * 16 + (lane & 15)) * BKP + ((lane >> 4) << 3);
#pragma unroll
    for (int nt = 0; nt < 4; ++nt)
        brow[nt] = (wn + nt * 16 + (lane & 15)) * BKP + ((lane >> 4) << 3);

    v8f acc[2][4];
#pragma unroll
    for (int mt = 0; mt < 2; ++mt)
#pragma unroll
        for (int nt = 0; nt < 4; ++nt)
#pragma unroll
            for (int k = 0; k < 8; ++k)
                acc[mt][nt][k] = 0.0f;

    // ---- prologue: stage 0 -> regs -> split-convert -> LDS buffer 0 ----
    float4 sa[4], sw[4];
#pragma unroll
    for (int i = 0; i < 4; ++i) {
        sa[i] = *(const float4*)(Ag + goff[i]);
        sw[i] = *(const float4*)(Wg + goff[i]);
    }
    split_store(sa, sw, Ahi[0], Alo[0], Whi[0], Wlo[0], loff);
    __syncthreads();

#pragma unroll 2
    for (int s = 0; s < NSTAGE; ++s) {
        const int cur = s & 1;
        const int nxt = cur ^ 1;

        // software pipeline: issue next stage's global loads before compute
        if (s + 1 < NSTAGE) {
            const float* ap = Ag + (s + 1) * BK;
            const float* wp = Wg + (s + 1) * BK;
#pragma unroll
            for (int i = 0; i < 4; ++i) {
                sa[i] = *(const float4*)(ap + goff[i]);
                sw[i] = *(const float4*)(wp + goff[i]);
            }
        }

        // ---- load WMMA fragments from LDS (ISA 7.12.2 wave32 layouts) ----
        v16bf afh[2], afl[2], bfh[4], bfl[4];
#pragma unroll
        for (int mt = 0; mt < 2; ++mt) {
            // A 16x32 bf16: lane holds row (lane&15); K runs of 8 at
            // krun=(lane>>4)*8 and krun+16.
            Frag32 fh, fl;
            fh.u[0] = *(const uint4*)(&Ahi[cur][arow[mt]]);
            fh.u[1] = *(const uint4*)(&Ahi[cur][arow[mt] + 16]);
            fl.u[0] = *(const uint4*)(&Alo[cur][arow[mt]]);
            fl.u[1] = *(const uint4*)(&Alo[cur][arow[mt] + 16]);
            afh[mt] = fh.v;
            afl[mt] = fl.v;
        }
#pragma unroll
        for (int nt = 0; nt < 4; ++nt) {
            // B 32x16 bf16: lane holds col (lane&15); K runs of 8 at
            // krun=(lane>>4)*8 and krun+16. w1 tile is [col][k] row-major already.
            Frag32 fh, fl;
            fh.u[0] = *(const uint4*)(&Whi[cur][brow[nt]]);
            fh.u[1] = *(const uint4*)(&Whi[cur][brow[nt] + 16]);
            fl.u[0] = *(const uint4*)(&Wlo[cur][brow[nt]]);
            fl.u[1] = *(const uint4*)(&Wlo[cur][brow[nt] + 16]);
            bfh[nt] = fh.v;
            bfl[nt] = fl.v;
        }

        // ---- bf16x3: hi*hi + hi*lo + lo*hi (lo*lo ~2^-16, dropped) ----
#pragma unroll
        for (int mt = 0; mt < 2; ++mt)
#pragma unroll
            for (int nt = 0; nt < 4; ++nt) {
                acc[mt][nt] = __builtin_amdgcn_wmma_f32_16x16x32_bf16(
                    false, afh[mt], false, bfh[nt], (short)0, acc[mt][nt], false, false);
                acc[mt][nt] = __builtin_amdgcn_wmma_f32_16x16x32_bf16(
                    false, afh[mt], false, bfl[nt], (short)0, acc[mt][nt], false, false);
                acc[mt][nt] = __builtin_amdgcn_wmma_f32_16x16x32_bf16(
                    false, afl[mt], false, bfh[nt], (short)0, acc[mt][nt], false, false);
            }

        // write next stage into the OTHER buffer: overlaps with compute of
        // `cur`, so only ONE barrier per stage is needed.
        if (s + 1 < NSTAGE)
            split_store(sa, sw, Ahi[nxt], Alo[nxt], Whi[nxt], Wlo[nxt], loff);
        __syncthreads();
    }

    // ---- epilogue: C/D layout VGPR i -> rows i (lanes 0-15) / i+8 (16-31) ----
    float* ob = out + (size_t)nz * (BLKSZ * BLKSZ);
#pragma unroll
    for (int mt = 0; mt < 2; ++mt)
#pragma unroll
        for (int nt = 0; nt < 4; ++nt) {
            const int col   = wn + nt * 16 + (lane & 15);
            const int rbase = wm + mt * 16 + ((lane >> 4) << 3);
#pragma unroll
            for (int i = 0; i < 8; ++i)
                ob[(rbase + i) * BLKSZ + col] = acc[mt][nt][i];
        }
}

extern "C" void kernel_launch(void* const* d_in, const int* in_sizes, int n_in,
                              void* d_out, int out_size, void* d_ws, size_t ws_size,
                              hipStream_t stream)
{
    const float* x  = (const float*)d_in[0];
    const float* w1 = (const float*)d_in[1];
    const int*   ri = (const int*)d_in[2];
    const int*   ci = (const int*)d_in[3];
    const int    nnz = in_sizes[2];   // 2048 nonzero 128x128 blocks

    sdd_bf16x3_kernel<<<nnz, 256, 0, stream>>>(x, w1, ri, ci, (float*)d_out);
}
